// VectorQuantizer_11467562680733
// MI455X (gfx1250) — matrix operand order repaired and checked by reading, not compiled, not run on hardware
//
#include <hip/hip_runtime.h>
#include <stdint.h>

// ---------------------------------------------------------------------------
// Types for CDNA5 WMMA (wave32): A/B = 16 bf16 (8 VGPRs), C/D = 8 f32.
// ---------------------------------------------------------------------------
typedef __bf16 v16bf __attribute__((ext_vector_type(16)));
typedef float  v8f   __attribute__((ext_vector_type(8)));

#define DDIM   256                 // latent / codeword dimension
#define KCB    1024                // number of codewords
#define NROWS  (8 * 8192)          // total latent vectors (B*N)
#define CTILES (KCB / 16)          // 64 codebook tiles of 16
#define OUTQ   ((size_t)NROWS * DDIM)

static __device__ __forceinline__ uint16_t bf16_bits(__bf16 h) {
    uint16_t u; __builtin_memcpy(&u, &h, 2); return u;
}

// Load 16 contiguous bf16 (32 B, aligned) into a v16bf.
static __device__ __forceinline__ v16bf load_bf16x16(const uint16_t* p) {
    union { uint4 q[2]; v16bf v; } u;
    u.q[0] = ((const uint4*)p)[0];
    u.q[1] = ((const uint4*)p)[1];
    return u.v;
}

static __device__ __forceinline__ v8f wmma_bf16(v16bf a, v16bf b, v8f c) {
    // 8 args: (neg_a, A, neg_b, B, c_mod, C, reuse_a, reuse_b)
    return __builtin_amdgcn_wmma_f32_16x16x32_bf16(false, a, false, b,
                                                   (short)0, c, false, false);
}

// ---------------------------------------------------------------------------
// Kernel 1: weights [1024,256] f32 -> bf16 hi/lo split + ||w||^2 per codeword
// ---------------------------------------------------------------------------
__global__ void vq_prep_weights(const float* __restrict__ weights,
                                uint16_t* __restrict__ whi,
                                uint16_t* __restrict__ wlo,
                                float* __restrict__ w2) {
    __shared__ float red[256];
    const int c = blockIdx.x;        // codeword
    const int t = threadIdx.x;       // dimension
    const size_t idx = (size_t)c * DDIM + t;
    float w   = weights[idx];
    __bf16 h  = (__bf16)w;
    float hf  = (float)h;
    __bf16 l  = (__bf16)(w - hf);
    whi[idx] = bf16_bits(h);
    wlo[idx] = bf16_bits(l);
    red[t] = w * w;
    __syncthreads();
    for (int s = 128; s > 0; s >>= 1) {
        if (t < s) red[t] += red[t + s];
        __syncthreads();
    }
    if (t == 0) w2[c] = red[0];
}

// ---------------------------------------------------------------------------
// Kernel 2: per-wave 16-row tile, sweep all 64 codebook tiles with
// split-bf16 WMMA, running argmin, cross-lane reduce, gather winner rows.
// Block = 256 threads = 8 waves = 128 rows; grid = 512.
// ---------------------------------------------------------------------------
__global__ __launch_bounds__(256)
void vq_main_kernel(const float* __restrict__ latents,
                    const uint16_t* __restrict__ whi,
                    const uint16_t* __restrict__ wlo,
                    const float* __restrict__ w2,
                    const float* __restrict__ weights,
                    float* __restrict__ out,
                    float* __restrict__ dotbuf) {
    const int lane  = threadIdx.x & 31;
    const int wv    = threadIdx.x >> 5;
    const int row0  = (blockIdx.x * 8 + wv) * 16;   // first of 16 rows
    const int mrow  = lane & 15;
    const int khalf = lane >> 4;

    // ---- Preload + convert the A tile (16 rows x 256 dims) into registers.
    // 16-bit A layout (16x32 per chunk): lanes hold row M = lane&15;
    // lane half khalf takes K = base + 16*khalf + 0..15 (contiguous, matches B).
    const float* xrow = latents + (size_t)(row0 + mrow) * DDIM;
    v16bf a_hi[8], a_lo[8];
#pragma unroll
    for (int kc = 0; kc < 8; ++kc) {
        const int kb = kc * 32;
        float va[16];
        const float4* p0 = (const float4*)(xrow + kb + 16 * khalf);
        const float4* p1 = (const float4*)(xrow + kb + 8 + 16 * khalf);
        float4 f0 = p0[0], f1 = p0[1], f2 = p1[0], f3 = p1[1];
        va[0]=f0.x; va[1]=f0.y; va[2]=f0.z; va[3]=f0.w;
        va[4]=f1.x; va[5]=f1.y; va[6]=f1.z; va[7]=f1.w;
        va[8]=f2.x; va[9]=f2.y; va[10]=f2.z; va[11]=f2.w;
        va[12]=f3.x; va[13]=f3.y; va[14]=f3.z; va[15]=f3.w;
#pragma unroll
        for (int e = 0; e < 16; ++e) {
            float f   = va[e];
            __bf16 h  = (__bf16)f;
            a_hi[kc][e] = h;
            a_lo[kc][e] = (__bf16)(f - (float)h);
        }
    }

    // ---- Sweep codebook tiles; keep running (min dist, index) per lane.
    float    vmin[8];
    uint32_t imin[8];
#pragma unroll
    for (int r = 0; r < 8; ++r) { vmin[r] = 3.402823466e38f; imin[r] = 0u; }

    for (int ct = 0; ct < CTILES; ++ct) {
        const int c = ct * 16 + mrow;               // this lane's column (N)
        const uint16_t* bh = whi + (size_t)c * DDIM + 16 * khalf;
        const uint16_t* bl = wlo + (size_t)c * DDIM + 16 * khalf;
        // gfx1250 prefetch of next tile's B data (stays inside d_ws).
        __builtin_prefetch(bh + 16 * DDIM, 0, 3);
        __builtin_prefetch(bl + 16 * DDIM, 0, 3);

        v8f acc0 = {}, acc1 = {};
#pragma unroll
        for (int kc = 0; kc < 8; kc += 2) {
            v16bf Bh0 = load_bf16x16(bh + (kc    ) * 32);
            v16bf Bl0 = load_bf16x16(bl + (kc    ) * 32);
            v16bf Bh1 = load_bf16x16(bh + (kc + 1) * 32);
            v16bf Bl1 = load_bf16x16(bl + (kc + 1) * 32);
            // 3-term split-precision product: hh + hl + lh (fp32 accumulate)
            acc0 = wmma_bf16(a_hi[kc    ], Bh0, acc0);
            acc1 = wmma_bf16(a_hi[kc + 1], Bh1, acc1);
            acc0 = wmma_bf16(a_hi[kc    ], Bl0, acc0);
            acc1 = wmma_bf16(a_hi[kc + 1], Bl1, acc1);
            acc0 = wmma_bf16(a_lo[kc    ], Bh0, acc0);
            acc1 = wmma_bf16(a_lo[kc + 1], Bh1, acc1);
        }
        const float w2c = w2[c];
#pragma unroll
        for (int r = 0; r < 8; ++r) {
            float s = w2c - 2.0f * (acc0[r] + acc1[r]);   // dist^2 - ||x||^2
            if (s < vmin[r]) { vmin[r] = s; imin[r] = (uint32_t)c; }
        }
    }

    // ---- Cross-lane argmin over the 16 lanes of each half-wave
    // (rows 0-7 live in lanes 0-15, rows 8-15 in lanes 16-31).
#pragma unroll
    for (int off = 1; off < 16; off <<= 1) {
#pragma unroll
        for (int r = 0; r < 8; ++r) {
            int addr = ((lane ^ off) << 2);
            float ov = __int_as_float(
                __builtin_amdgcn_ds_bpermute(addr, __float_as_int(vmin[r])));
            uint32_t oi = (uint32_t)
                __builtin_amdgcn_ds_bpermute(addr, (int)imin[r]);
            if (ov < vmin[r] || (ov == vmin[r] && oi < imin[r])) {
                vmin[r] = ov; imin[r] = oi;
            }
        }
    }

    // ---- Winner dot products (for the softmax loss): dot = (w2 - dist)/2.
    if (mrow == 0) {
#pragma unroll
        for (int r = 0; r < 8; ++r) {
            int m = row0 + khalf * 8 + r;
            dotbuf[m] = 0.5f * (w2[imin[r]] - vmin[r]);
        }
    }

    // ---- Gather: all 32 lanes copy each winning fp32 codeword row (1 KB).
#pragma unroll 4
    for (int mi = 0; mi < 16; ++mi) {
        const int srcLane = (mi < 8) ? 0 : 16;
        const int r = mi & 7;
        uint32_t cstar =
            (uint32_t)__builtin_amdgcn_readlane((int)imin[r], srcLane);
        const float4* src = (const float4*)(weights + (size_t)cstar * DDIM);
        float4* dst = (float4*)(out + (size_t)(row0 + mi) * DDIM);
        dst[lane]      = src[lane];        // dims   0..127
        dst[lane + 32] = src[lane + 32];   // dims 128..255
    }
}

// ---------------------------------------------------------------------------
// Kernel 3: per-batch sum of softmax over N=8192 winner dots (deterministic).
// ---------------------------------------------------------------------------
__global__ void vq_loss_partial(const float* __restrict__ dotbuf,
                                float* __restrict__ partial) {
    __shared__ float red[256];
    const int b = blockIdx.x, t = threadIdx.x;
    const float* x = dotbuf + (size_t)b * 8192;
    float mx = -3.402823466e38f;
    for (int i = t; i < 8192; i += 256) mx = fmaxf(mx, x[i]);
    red[t] = mx; __syncthreads();
    for (int s = 128; s > 0; s >>= 1) {
        if (t < s) red[t] = fmaxf(red[t], red[t + s]);
        __syncthreads();
    }
    mx = red[0]; __syncthreads();
    float se = 0.0f;
    for (int i = t; i < 8192; i += 256) se += __expf(x[i] - mx);
    red[t] = se; __syncthreads();
    for (int s = 128; s > 0; s >>= 1) {
        if (t < s) red[t] += red[t + s];
        __syncthreads();
    }
    const float inv = 1.0f / red[0]; __syncthreads();
    float ls = 0.0f;
    for (int i = t; i < 8192; i += 256) ls += __expf(x[i] - mx) * inv;
    red[t] = ls; __syncthreads();
    for (int s = 128; s > 0; s >>= 1) {
        if (t < s) red[t] += red[t + s];
        __syncthreads();
    }
    if (t == 0) partial[b] = red[0];
}

__global__ void vq_loss_final(const float* __restrict__ partial,
                              float* __restrict__ out_loss) {
    float s = 0.0f;
#pragma unroll
    for (int b = 0; b < 8; ++b) s += partial[b];
    out_loss[0] = s;
}

// ---------------------------------------------------------------------------
// Launch: prep -> main GEMM/argmin/gather -> loss partials -> loss sum.
// Workspace layout (bytes): whi 512K | wlo 512K | w2 4K | dotbuf 256K | part 32
// ---------------------------------------------------------------------------
extern "C" void kernel_launch(void* const* d_in, const int* in_sizes, int n_in,
                              void* d_out, int out_size, void* d_ws, size_t ws_size,
                              hipStream_t stream) {
    const float* latents = (const float*)d_in[0];   // [8,8192,256] f32
    const float* weights = (const float*)d_in[1];   // [1024,256]   f32
    float* out = (float*)d_out;                     // quantized | loss

    char* ws = (char*)d_ws;
    uint16_t* whi    = (uint16_t*)(ws);
    uint16_t* wlo    = (uint16_t*)(ws + (size_t)512 * 1024);
    float*    w2     = (float*)   (ws + (size_t)1024 * 1024);
    float*    dotbuf = (float*)   (ws + (size_t)1024 * 1024 + 4096);
    float*    part   = (float*)   (ws + (size_t)1024 * 1024 + 4096 + 262144);

    vq_prep_weights<<<KCB, 256, 0, stream>>>(weights, whi, wlo, w2);
    vq_main_kernel<<<NROWS / 128, 256, 0, stream>>>(latents, whi, wlo, w2,
                                                    weights, out, dotbuf);
    vq_loss_partial<<<8, 256, 0, stream>>>(dotbuf, part);
    vq_loss_final<<<1, 1, 0, stream>>>(part, out + OUTQ);
}
